// GRC_Self_Attention_16174846836954
// MI455X (gfx1250) — matrix operand order repaired and checked by reading, not compiled, not run on hardware
//
#include <hip/hip_runtime.h>
#include <math.h>
#include <stdint.h>

typedef __attribute__((ext_vector_type(16))) _Float16 v16h;
typedef __attribute__((ext_vector_type(8)))  _Float16 v8h;
typedef __attribute__((ext_vector_type(8)))  float    v8f;
typedef __attribute__((ext_vector_type(4)))  int      v4i;

#define B_     8
#define T_     1025
#define C_     512
#define HEADS_ 8
#define CACHE_ 256
#define L_     256
#define TP_    1056            // T rounded up so every key/query tile stays in-buffer
#define NTP_   (B_ * TP_)      // 8448, multiple of 64 -> no GEMM M guards

#if defined(__has_builtin)
#if __has_builtin(__builtin_amdgcn_global_load_async_to_lds_b128)
#define HAVE_ASYNC_LDS 1
#endif
#endif
#ifndef HAVE_ASYNC_LDS
#define HAVE_ASYNC_LDS 0
#endif

#if HAVE_ASYNC_LDS
typedef __attribute__((address_space(1))) v4i* gv4i_p;
typedef __attribute__((address_space(3))) v4i* lv4i_p;
#endif

__device__ inline v16h mk16(v8h lo, v8h hi) {
  v16h r;
#pragma unroll
  for (int i = 0; i < 8; ++i) { r[i] = lo[i]; r[i + 8] = hi[i]; }
  return r;
}

// ---------------------------------------------------------------------------
// WMMA GEMM: C[M,N] = act(A[M,K] @ W[K,N] + bias). A,W f16; f32 accumulate.
// Block = 256 threads = 8 waves (2x4); block tile 64x128; each wave owns a
// 32x32 register tile (2x2 WMMA accumulators) -> 4 v_wmma per K-step with
// 2x operand reuse on both A and B fragments. A tile is staged to LDS via
// GLOBAL_LOAD_ASYNC_TO_LDS_B128 (ASYNCcnt) when the builtin is available.
// ---------------------------------------------------------------------------
__global__ __launch_bounds__(256)
void wmma_gemm(const _Float16* __restrict__ A, int lda,
               const _Float16* __restrict__ Bw, int ldb,
               const float* __restrict__ bias,
               float* __restrict__ Cf32, _Float16* __restrict__ Cf16, int ldc,
               int M, int N, int K, int act) {
  __shared__ alignas(16) _Float16 sA[64][40];
  __shared__ alignas(16) _Float16 sB[128][40];   // sB[n][k] (W transposed)
  const int tid  = threadIdx.x;
  const int lane = tid & 31, w = tid >> 5;
  const int lid  = lane & 15, grp = lane >> 4;
  const int m0 = blockIdx.y * 64, n0 = blockIdx.x * 128;
  const int wm = (w >> 2) * 32, wn = (w & 3) * 32;
  (void)M; (void)N;

  v8f acc[2][2];
#pragma unroll
  for (int mi = 0; mi < 2; ++mi)
#pragma unroll
    for (int ni = 0; ni < 2; ++ni)
#pragma unroll
      for (int i = 0; i < 8; ++i) acc[mi][ni][i] = 0.f;

  const int ra = tid >> 2, ca = (tid & 3) << 3;   // A tile: 8 halves/thread
  const int bbase = tid << 4;                     // B tile: 16 halves/thread
  const int bn = bbase >> 5, bk = bbase & 31;     // bk in {0,16}

  for (int k0 = 0; k0 < K; k0 += 32) {
    const _Float16* gA = A + (size_t)(m0 + ra) * lda + k0 + ca;
#if HAVE_ASYNC_LDS
    // async global->LDS, no VGPR round trip; tracked by ASYNCcnt
    __builtin_amdgcn_global_load_async_to_lds_b128(
        (gv4i_p)(uintptr_t)(const void*)gA,
        (lv4i_p)(uint32_t)(uintptr_t)(void*)&sA[ra][ca],
        0, 0);
#else
    *(v8h*)&sA[ra][ca] = *(const v8h*)gA;
#endif
#pragma unroll
    for (int i = 0; i < 16; ++i)
      sB[bn][bk + i] = Bw[(size_t)(k0 + bk + i) * ldb + n0 + bn];
    __builtin_prefetch(A + (size_t)(m0 + ra) * lda + k0 + 32 + ca, 0, 3);
#if HAVE_ASYNC_LDS
    asm volatile("s_wait_asynccnt 0" ::: "memory");
#endif
    __syncthreads();

    // A fragments (16x32): lanes 0-15 -> K 0..7 & 16..23; lanes 16-31 -> +8
    const int akb = grp ? 8 : 0;
    v16h af[2];
    af[0] = mk16(*(const v8h*)&sA[wm + lid][akb],
                 *(const v8h*)&sA[wm + lid][akb + 16]);
    af[1] = mk16(*(const v8h*)&sA[wm + 16 + lid][akb],
                 *(const v8h*)&sA[wm + 16 + lid][akb + 16]);
    // B fragments (32x16): lanes 0-15 -> K 0..7 & 16..23; lanes 16-31 -> +8
    const int bkb = grp ? 8 : 0;
    v16h bf[2];
    bf[0] = mk16(*(const v8h*)&sB[wn + lid][bkb],
                 *(const v8h*)&sB[wn + lid][bkb + 16]);
    bf[1] = mk16(*(const v8h*)&sB[wn + 16 + lid][bkb],
                 *(const v8h*)&sB[wn + 16 + lid][bkb + 16]);

#pragma unroll
    for (int mi = 0; mi < 2; ++mi)
#pragma unroll
      for (int ni = 0; ni < 2; ++ni)
        acc[mi][ni] = __builtin_amdgcn_wmma_f32_16x16x32_f16(
            false, af[mi], false, bf[ni], (short)0, acc[mi][ni], false, false);
    __syncthreads();
  }

#pragma unroll
  for (int ni = 0; ni < 2; ++ni) {
    const int col = n0 + wn + ni * 16 + lid;
    const float bv = bias ? bias[col] : 0.f;
#pragma unroll
    for (int mi = 0; mi < 2; ++mi)
#pragma unroll
      for (int i = 0; i < 8; ++i) {
        const int row = m0 + wm + mi * 16 + grp * 8 + i;
        float val = acc[mi][ni][i] + bv;
        if (act == 1)      val = 1.f / (1.f + __expf(-val));                       // sigmoid
        else if (act == 2) val = 0.5f * val * (1.f + erff(val * 0.70710678118f));  // exact gelu
        if (Cf32) Cf32[(size_t)row * ldc + col] = val;
        if (Cf16) Cf16[(size_t)row * ldc + col] = (_Float16)val;
      }
  }
}

// ---------------------------------------------------------------------------
// Flash attention, one wave per 16-row query tile of one (b,h).
// S = Q K^T via WMMA (K^T B-fragments load contiguous from global since the
// contraction runs along dh). Online softmax with 16-lane shfl_xor reductions.
// P goes through LDS into A-fragment layout; V is LDS-transposed for P·V.
// ---------------------------------------------------------------------------
template <int DH>
__global__ __launch_bounds__(32)
void flash_attn(const _Float16* __restrict__ q, const _Float16* __restrict__ k,
                const _Float16* __restrict__ v, _Float16* __restrict__ o,
                int Tk, int ldq, int ldkv, int ldo,
                int TPq, int TPk, float scale) {
  __shared__ alignas(16) _Float16 pLDS[16][40];
  __shared__ alignas(16) _Float16 vLDS[DH][40];
  const int lane = threadIdx.x;
  const int lid = lane & 15, grp = lane >> 4;
  const int qt0 = blockIdx.x * 16;
  const int h = blockIdx.y, b = blockIdx.z;

  const _Float16* Q = q + (size_t)b * TPq * ldq  + (size_t)h * DH;
  const _Float16* K = k + (size_t)b * TPk * ldkv + (size_t)h * DH;
  const _Float16* V = v + (size_t)b * TPk * ldkv + (size_t)h * DH;
  _Float16*       O = o + (size_t)b * TPq * ldo  + (size_t)h * DH;

  // Preload Q A-fragments (16 x 32 per dh-chunk)
  v16h qf[DH / 32];
  {
    const int qrow = qt0 + lid;
    const int kb = grp ? 8 : 0;
#pragma unroll
    for (int j = 0; j < DH / 32; ++j) {
      const _Float16* qp = Q + (size_t)qrow * ldq + j * 32 + kb;
      qf[j] = mk16(*(const v8h*)qp, *(const v8h*)(qp + 16));
    }
  }

  float mrow[8], lrow[8];
  v8f oacc[DH / 16];
#pragma unroll
  for (int i = 0; i < 8; ++i) { mrow[i] = -1e30f; lrow[i] = 0.f; }
#pragma unroll
  for (int n = 0; n < DH / 16; ++n)
#pragma unroll
    for (int i = 0; i < 8; ++i) oacc[n][i] = 0.f;

  const int nkt = (Tk + 31) >> 5;
  for (int kt = 0; kt < nkt; ++kt) {
    const int kt0 = kt << 5;

    // ---- scores: two 16x16 S tiles covering 32 keys ----
    v8f s0, s1;
#pragma unroll
    for (int i = 0; i < 8; ++i) { s0[i] = 0.f; s1[i] = 0.f; }
#pragma unroll
    for (int j = 0; j < DH / 32; ++j) {
      const int dh0 = j * 32 + (grp ? 8 : 0);
      const _Float16* kp0 = K + (size_t)(kt0 + lid) * ldkv + dh0;
      v16h kf0 = mk16(*(const v8h*)kp0, *(const v8h*)(kp0 + 16));
      s0 = __builtin_amdgcn_wmma_f32_16x16x32_f16(false, qf[j], false, kf0,
                                                  (short)0, s0, false, false);
      const _Float16* kp1 = K + (size_t)(kt0 + 16 + lid) * ldkv + dh0;
      v16h kf1 = mk16(*(const v8h*)kp1, *(const v8h*)(kp1 + 16));
      s1 = __builtin_amdgcn_wmma_f32_16x16x32_f16(false, qf[j], false, kf1,
                                                  (short)0, s1, false, false);
    }

    // ---- stage V^T tile (vLDS[dh][key_local]) ----
    {
      const _Float16* vp = V + (size_t)(kt0 + lane) * ldkv;
#pragma unroll
      for (int c8 = 0; c8 < DH; c8 += 8) {
        v8h vv = *(const v8h*)(vp + c8);
#pragma unroll
        for (int e = 0; e < 8; ++e) vLDS[c8 + e][lane] = vv[e];
      }
    }

    const bool ok0 = (kt0 + lid)      < Tk;
    const bool ok1 = (kt0 + 16 + lid) < Tk;

    // ---- online softmax over this 32-key chunk ----
#pragma unroll
    for (int i = 0; i < 8; ++i) {
      float a0 = ok0 ? s0[i] * scale : -1e30f;
      float a1 = ok1 ? s1[i] * scale : -1e30f;
      float rm = fmaxf(a0, a1);
#pragma unroll
      for (int ms = 1; ms < 16; ms <<= 1) rm = fmaxf(rm, __shfl_xor(rm, ms, 32));
      const float mnew = fmaxf(mrow[i], rm);
      const float corr = __expf(mrow[i] - mnew);
      const float p0 = __expf(a0 - mnew);
      const float p1 = __expf(a1 - mnew);
      float rs = p0 + p1;
#pragma unroll
      for (int ms = 1; ms < 16; ms <<= 1) rs += __shfl_xor(rs, ms, 32);
      lrow[i] = lrow[i] * corr + rs;
      mrow[i] = mnew;
#pragma unroll
      for (int n = 0; n < DH / 16; ++n) oacc[n][i] *= corr;
      pLDS[grp * 8 + i][lid]      = (_Float16)p0;
      pLDS[grp * 8 + i][16 + lid] = (_Float16)p1;
    }
    asm volatile("s_wait_dscnt 0" ::: "memory");   // same-wave LDS RAW fence

    // ---- O += P(16x32) @ V(32xDH) ----
    v16h pf;
    {
      const int kb = grp ? 8 : 0;
      pf = mk16(*(const v8h*)&pLDS[lid][kb], *(const v8h*)&pLDS[lid][kb + 16]);
    }
#pragma unroll
    for (int n = 0; n < DH / 16; ++n) {
      const int bkb = grp ? 8 : 0;
      v16h vf = mk16(*(const v8h*)&vLDS[n * 16 + lid][bkb],
                     *(const v8h*)&vLDS[n * 16 + lid][bkb + 16]);
      oacc[n] = __builtin_amdgcn_wmma_f32_16x16x32_f16(false, pf, false, vf,
                                                       (short)0, oacc[n], false, false);
    }
    __builtin_prefetch(K + (size_t)(kt0 + 32 + lid) * ldkv, 0, 3);
  }

  // ---- epilogue: O = acc / l ----
#pragma unroll
  for (int n = 0; n < DH / 16; ++n)
#pragma unroll
    for (int i = 0; i < 8; ++i) {
      const int row = qt0 + grp * 8 + i;
      O[(size_t)row * ldo + n * 16 + lid] = (_Float16)(oacc[n][i] / lrow[i]);
    }
}

// ---------------------------------------------------------------------------
// Elementwise helpers
// ---------------------------------------------------------------------------
__global__ void cvt_f32_f16(const float* __restrict__ s, _Float16* __restrict__ d, int n) {
  int i = blockIdx.x * blockDim.x + threadIdx.x;
  if (i < n) d[i] = (_Float16)s[i];
}

// x (B,T,C) f32 -> x16 (B,TP,C) f16, zero-filling the pad rows
__global__ void cvt_x_pad(const float* __restrict__ x, _Float16* __restrict__ x16) {
  int i = blockIdx.x * blockDim.x + threadIdx.x;
  if (i >= NTP_ * C_) return;
  int c = i & (C_ - 1);
  int r = (i >> 9) % TP_;
  int b = i / (TP_ * C_);
  x16[i] = (r < T_) ? (_Float16)x[((size_t)b * T_ + r) * C_ + c] : (_Float16)0.f;
}

// cat = [gr_cache (bcast), x_summary] as f16, rows = B*L, cols = 512
__global__ void build_cat(const float* __restrict__ grc, const float* __restrict__ xself,
                          _Float16* __restrict__ cat) {
  int i = blockIdx.x * blockDim.x + threadIdx.x;
  if (i >= B_ * L_ * 512) return;
  int col = i & 511, row = i >> 9;
  int l = row & (L_ - 1), b = row >> 8;
  if (col < CACHE_) {
    cat[i] = (_Float16)grc[l * CACHE_ + col];
  } else {
    int tsel = (l * T_) >> 8;  // arange(L)*T//L
    cat[i] = (_Float16)xself[((size_t)b * TP_ + tsel) * C_ + (col - CACHE_)];
  }
}

// cat2 = [reset*gr, x_summary]
__global__ void build_cat2(const float* __restrict__ rbuf, const float* __restrict__ grc,
                           const _Float16* __restrict__ cat, _Float16* __restrict__ cat2) {
  int i = blockIdx.x * blockDim.x + threadIdx.x;
  if (i >= B_ * L_ * 512) return;
  int col = i & 511, row = i >> 9;
  int l = row & (L_ - 1);
  cat2[i] = (col < CACHE_)
              ? (_Float16)(rbuf[(size_t)row * CACHE_ + col] * grc[l * CACHE_ + col])
              : cat[i];
}

// layernorm(add) then gr_new = z*ln + (1-z)*gr  (f16 out for the KV GEMMs)
__global__ __launch_bounds__(256)
void gru_finish(const float* __restrict__ addb, const float* __restrict__ zbuf,
                const float* __restrict__ grc, const float* __restrict__ gamma,
                const float* __restrict__ beta, _Float16* __restrict__ grn) {
  __shared__ float red[256];
  const int row = blockIdx.x, tid = threadIdx.x;
  const int l = row & (L_ - 1);
  const float val = addb[(size_t)row * CACHE_ + tid];
  red[tid] = val; __syncthreads();
  for (int s = 128; s > 0; s >>= 1) { if (tid < s) red[tid] += red[tid + s]; __syncthreads(); }
  const float mu = red[0] / 256.f; __syncthreads();
  const float d = val - mu;
  red[tid] = d * d; __syncthreads();
  for (int s = 128; s > 0; s >>= 1) { if (tid < s) red[tid] += red[tid + s]; __syncthreads(); }
  const float var = red[0] / 256.f;
  const float ln = d * rsqrtf(var + 1e-5f) * gamma[tid] + beta[tid];
  const float z = zbuf[(size_t)row * CACHE_ + tid];
  grn[(size_t)row * CACHE_ + tid] = (_Float16)(z * ln + (1.f - z) * grc[l * CACHE_ + tid]);
}

// out = alpha*[x_mem,0] + (1-alpha)*x_self + pos(conv), fused grouped 3x3 conv
__global__ void final_assemble(const float* __restrict__ x, const float* __restrict__ xself,
                               const float* __restrict__ xmem, const float* __restrict__ lam,
                               const float* __restrict__ conv_w, const float* __restrict__ conv_b,
                               float* __restrict__ out) {
  int idx = blockIdx.x * blockDim.x + threadIdx.x;
  if (idx >= B_ * T_ * C_) return;
  const int c = idx & (C_ - 1);
  const int t = (idx / C_) % T_;
  const int b = idx / (C_ * T_);
  const int h = c >> 6, d = c & 63;
  const float alpha = 1.f / (1.f + __expf(-lam[h]));
  const float xs = xself[((size_t)b * TP_ + t) * C_ + c];
  const float xm = (d < 32) ? xmem[((size_t)b * TP_ + t) * CACHE_ + h * 32 + d] : 0.f;
  float pos = 0.f;
  if (t > 0) {
    const int p = t - 1, ii = p >> 5, jj = p & 31;
    const int g2 = (c >> 1) << 1;  // 2 input channels per group
    float a = conv_b[c];
    for (int icl = 0; icl < 2; ++icl) {
      const int ic = g2 + icl;
      for (int kh = 0; kh < 3; ++kh) {
        const int i2 = ii + kh - 1;
        if (i2 < 0 || i2 >= 32) continue;
        for (int kw = 0; kw < 3; ++kw) {
          const int j2 = jj + kw - 1;
          if (j2 < 0 || j2 >= 32) continue;
          a += x[((size_t)b * T_ + 1 + i2 * 32 + j2) * C_ + ic] *
               conv_w[((c * 2 + icl) * 3 + kh) * 3 + kw];
        }
      }
    }
    pos = a;
  }
  out[idx] = alpha * xm + (1.f - alpha) * xs + pos;
}

// ---------------------------------------------------------------------------
extern "C" void kernel_launch(void* const* d_in, const int* in_sizes, int n_in,
                              void* d_out, int out_size, void* d_ws, size_t ws_size,
                              hipStream_t stream) {
  (void)in_sizes; (void)n_in; (void)out_size; (void)ws_size;
  const float* x      = (const float*)d_in[0];
  const float* grc    = (const float*)d_in[1];
  const float* Wq1 = (const float*)d_in[2];  const float* bq1 = (const float*)d_in[3];
  const float* Wk1 = (const float*)d_in[4];  const float* bk1 = (const float*)d_in[5];
  const float* Wv1 = (const float*)d_in[6];  const float* bv1 = (const float*)d_in[7];
  const float* Wo1 = (const float*)d_in[8];  const float* bo1 = (const float*)d_in[9];
  const float* Wq2 = (const float*)d_in[10]; const float* bq2 = (const float*)d_in[11];
  const float* Wk2 = (const float*)d_in[12]; const float* bk2 = (const float*)d_in[13];
  const float* Wv2 = (const float*)d_in[14]; const float* bv2 = (const float*)d_in[15];
  const float* Wo2 = (const float*)d_in[16]; const float* bo2 = (const float*)d_in[17];
  const float* Wr  = (const float*)d_in[18]; const float* br  = (const float*)d_in[19];
  const float* Wz  = (const float*)d_in[20]; const float* bz  = (const float*)d_in[21];
  const float* Wa  = (const float*)d_in[22]; const float* ba  = (const float*)d_in[23];
  const float* gamma = (const float*)d_in[24];
  const float* beta  = (const float*)d_in[25];
  const float* lam   = (const float*)d_in[26];
  const float* convw = (const float*)d_in[27];
  const float* convb = (const float*)d_in[28];
  float* out = (float*)d_out;

  // ---- workspace bump allocator (256B aligned) ----
  char* wp = (char*)d_ws;
  auto alloc = [&](size_t bytes) -> void* {
    void* r = (void*)wp;
    wp += (bytes + 255) & ~(size_t)255;
    return r;
  };
  _Float16* x16  = (_Float16*)alloc((size_t)NTP_ * C_ * 2);
  _Float16* wq1h = (_Float16*)alloc(512 * 512 * 2);
  _Float16* wk1h = (_Float16*)alloc(512 * 512 * 2);
  _Float16* wv1h = (_Float16*)alloc(512 * 512 * 2);
  _Float16* wo1h = (_Float16*)alloc(512 * 512 * 2);
  _Float16* wq2h = (_Float16*)alloc(256 * 256 * 2);
  _Float16* wk2h = (_Float16*)alloc(256 * 256 * 2);
  _Float16* wv2h = (_Float16*)alloc(256 * 256 * 2);
  _Float16* wo2h = (_Float16*)alloc(256 * 256 * 2);
  _Float16* wrh  = (_Float16*)alloc(512 * 256 * 2);
  _Float16* wzh  = (_Float16*)alloc(512 * 256 * 2);
  _Float16* wah  = (_Float16*)alloc(512 * 256 * 2);
  _Float16* q1   = (_Float16*)alloc((size_t)NTP_ * C_ * 2);
  _Float16* k1   = (_Float16*)alloc((size_t)NTP_ * C_ * 2);
  _Float16* v1   = (_Float16*)alloc((size_t)NTP_ * C_ * 2);
  _Float16* o1   = (_Float16*)alloc((size_t)NTP_ * C_ * 2);
  float*    xself= (float*)alloc((size_t)NTP_ * C_ * 4);
  _Float16* cat16= (_Float16*)alloc((size_t)B_ * L_ * 512 * 2);
  _Float16* cat2 = (_Float16*)alloc((size_t)B_ * L_ * 512 * 2);
  float*    rbuf = (float*)alloc((size_t)B_ * L_ * CACHE_ * 4);
  float*    zbuf = (float*)alloc((size_t)B_ * L_ * CACHE_ * 4);
  float*    addb = (float*)alloc((size_t)B_ * L_ * CACHE_ * 4);
  _Float16* grn  = (_Float16*)alloc((size_t)B_ * L_ * CACHE_ * 2);
  _Float16* q2   = (_Float16*)alloc((size_t)NTP_ * CACHE_ * 2);
  _Float16* k2   = (_Float16*)alloc((size_t)B_ * L_ * CACHE_ * 2);
  _Float16* v2   = (_Float16*)alloc((size_t)B_ * L_ * CACHE_ * 2);
  _Float16* o2   = (_Float16*)alloc((size_t)NTP_ * CACHE_ * 2);
  float*    xmem = (float*)alloc((size_t)NTP_ * CACHE_ * 4);

  const int TB = 256;
  auto g1 = [&](int n) { return dim3((n + TB - 1) / TB); };

  // ---- f32 -> f16 conversions ----
  cvt_x_pad<<<g1(NTP_ * C_), TB, 0, stream>>>(x, x16);
  cvt_f32_f16<<<g1(512 * 512), TB, 0, stream>>>(Wq1, wq1h, 512 * 512);
  cvt_f32_f16<<<g1(512 * 512), TB, 0, stream>>>(Wk1, wk1h, 512 * 512);
  cvt_f32_f16<<<g1(512 * 512), TB, 0, stream>>>(Wv1, wv1h, 512 * 512);
  cvt_f32_f16<<<g1(512 * 512), TB, 0, stream>>>(Wo1, wo1h, 512 * 512);
  cvt_f32_f16<<<g1(256 * 256), TB, 0, stream>>>(Wq2, wq2h, 256 * 256);
  cvt_f32_f16<<<g1(256 * 256), TB, 0, stream>>>(Wk2, wk2h, 256 * 256);
  cvt_f32_f16<<<g1(256 * 256), TB, 0, stream>>>(Wv2, wv2h, 256 * 256);
  cvt_f32_f16<<<g1(256 * 256), TB, 0, stream>>>(Wo2, wo2h, 256 * 256);
  cvt_f32_f16<<<g1(512 * 256), TB, 0, stream>>>(Wr, wrh, 512 * 256);
  cvt_f32_f16<<<g1(512 * 256), TB, 0, stream>>>(Wz, wzh, 512 * 256);
  cvt_f32_f16<<<g1(512 * 256), TB, 0, stream>>>(Wa, wah, 512 * 256);

  const dim3 gN512(C_ / 128, NTP_ / 64);          // (4,132)
  const dim3 gN256big(CACHE_ / 128, NTP_ / 64);   // (2,132)
  const dim3 gGru(CACHE_ / 128, (B_ * L_) / 64);  // (2,32)

  // ---- self attention ----
  wmma_gemm<<<gN512, 256, 0, stream>>>(x16, C_, wq1h, C_, bq1, nullptr, q1, C_, NTP_, C_, C_, 0);
  wmma_gemm<<<gN512, 256, 0, stream>>>(x16, C_, wk1h, C_, bk1, nullptr, k1, C_, NTP_, C_, C_, 0);
  wmma_gemm<<<gN512, 256, 0, stream>>>(x16, C_, wv1h, C_, bv1, nullptr, v1, C_, NTP_, C_, C_, 0);
  flash_attn<64><<<dim3(TP_ / 16, HEADS_, B_), 32, 0, stream>>>(
      q1, k1, v1, o1, T_, C_, C_, C_, TP_, TP_, 0.125f);
  wmma_gemm<<<gN512, 256, 0, stream>>>(o1, C_, wo1h, C_, bo1, xself, nullptr, C_, NTP_, C_, C_, 0);

  // ---- GRU cache update ----
  build_cat<<<g1(B_ * L_ * 512), TB, 0, stream>>>(grc, xself, cat16);
  wmma_gemm<<<gGru, 256, 0, stream>>>(cat16, 512, wrh, CACHE_, br, rbuf, nullptr, CACHE_,
                                      B_ * L_, CACHE_, 512, 1);
  wmma_gemm<<<gGru, 256, 0, stream>>>(cat16, 512, wzh, CACHE_, bz, zbuf, nullptr, CACHE_,
                                      B_ * L_, CACHE_, 512, 1);
  build_cat2<<<g1(B_ * L_ * 512), TB, 0, stream>>>(rbuf, grc, cat16, cat2);
  wmma_gemm<<<gGru, 256, 0, stream>>>(cat2, 512, wah, CACHE_, ba, addb, nullptr, CACHE_,
                                      B_ * L_, CACHE_, 512, 2);
  gru_finish<<<B_ * L_, 256, 0, stream>>>(addb, zbuf, grc, gamma, beta, grn);

  // ---- memory cross-attention ----
  wmma_gemm<<<gN256big, 256, 0, stream>>>(x16, C_, wq2h, CACHE_, bq2, nullptr, q2, CACHE_,
                                          NTP_, CACHE_, CACHE_, 0);   // uses x[:, :, :256]
  wmma_gemm<<<gGru, 256, 0, stream>>>(grn, CACHE_, wk2h, CACHE_, bk2, nullptr, k2, CACHE_,
                                      B_ * L_, CACHE_, CACHE_, 0);
  wmma_gemm<<<gGru, 256, 0, stream>>>(grn, CACHE_, wv2h, CACHE_, bv2, nullptr, v2, CACHE_,
                                      B_ * L_, CACHE_, CACHE_, 0);
  flash_attn<32><<<dim3(TP_ / 16, HEADS_, B_), 32, 0, stream>>>(
      q2, k2, v2, o2, L_, CACHE_, CACHE_, CACHE_, TP_, L_, 0.17677669529f);
  wmma_gemm<<<gN256big, 256, 0, stream>>>(o2, CACHE_, wo2h, CACHE_, bo2, xmem, nullptr, CACHE_,
                                          NTP_, CACHE_, CACHE_, 0);

  // ---- fused conv + mix ----
  final_assemble<<<g1(B_ * T_ * C_), TB, 0, stream>>>(x, xself, xmem, lam, convw, convb, out);
}
